// HWFriendlySNN_26886495273023
// MI455X (gfx1250) — compile-verified, matrix operand order checked by reading, not run on hardware
//
#include <hip/hip_runtime.h>

typedef _Float16 v8h  __attribute__((ext_vector_type(8)));
typedef _Float16 v16h __attribute__((ext_vector_type(16)));
typedef float    v8f  __attribute__((ext_vector_type(8)));

#define BETA   0.5f
#define THR    1.0f
#define NSTEP  16
#define KREAL  1352          // 8*13*13 pooled features
#define NKB    43            // ceil(1352/32)
#define KPAD   (NKB * 32)    // 1376

// ---------------------------------------------------------------------------
// Prep: split fc_w (f32, [10,1352]) into hi/lo f16 planes, padded to [16,1376]
// (rows 10..15 and cols 1352..1375 zero). w == (float)hi + (float)lo to ~2^-22.
// ---------------------------------------------------------------------------
__global__ void snn_prep_fcw(const float* __restrict__ fc_w,
                             _Float16* __restrict__ hi,
                             _Float16* __restrict__ lo) {
    int idx = blockIdx.x * blockDim.x + threadIdx.x;
    if (idx >= 16 * KPAD) return;
    int n = idx / KPAD;
    int k = idx - n * KPAD;
    float v = (n < 10 && k < KREAL) ? fc_w[n * KREAL + k] : 0.0f;
    _Float16 h = (_Float16)v;
    hi[idx] = h;
    lo[idx] = (_Float16)(v - (float)h);
}

// ---------------------------------------------------------------------------
// Fully fused SNN: one block (256 threads = 8 waves) per image.
//   Phase 1: conv once -> cur1 in regs; 16-step LIF-1 + 2x2 avgpool -> LDS f16
//   Phase 2: [16 x 1376] x [1376 x 16] GEMM via v_wmma_f32_16x16x32_f16,
//            K-blocks split over 8 waves (scalarized wave index so the WMMA
//            loop uses SALU control flow and EXEC stays all-ones),
//            hi+lo weight planes accumulated into one f32 accumulator
//   Phase 3: deterministic fixed-order sum of 8 wave partials, then serial mem2 LIF + spike count
// ---------------------------------------------------------------------------
__global__ __launch_bounds__(256) void snn_fused(
    const float* __restrict__ x, const float* __restrict__ conv_w,
    const _Float16* __restrict__ fhi, const _Float16* __restrict__ flo,
    float* __restrict__ out)
{
    __shared__ float xs[784];                       // 28x28 input image
    __shared__ float ws[72];                        // 8x1x3x3 conv weights
    __shared__ __align__(16) _Float16 pool[NSTEP * KPAD]; // 16 steps of pooled spikes
    __shared__ float partial[8 * 256];              // per-wave WMMA partials
    __shared__ float cur2[16 * 16];                 // [timestep][neuron]

    const int tid = threadIdx.x;
    const int b   = blockIdx.x;
    const float* xb = x + b * 784;

    for (int i = tid; i < 784; i += 256) xs[i] = xb[i];
    for (int i = tid; i < 72;  i += 256) ws[i] = conv_w[i];
    // zero the K padding columns once (all 16 rows, k in [1352,1376))
    for (int i = tid; i < NSTEP * (KPAD - KREAL); i += 256) {
        int t = i / (KPAD - KREAL);
        int k = KREAL + (i - t * (KPAD - KREAL));
        pool[t * KPAD + k] = (_Float16)0.0f;
    }
    __syncthreads();

    // ---- Phase 1: conv + LIF layer 1 + avgpool ----
    // Each thread owns up to 6 pool cells (idx = tid + i*256 < 1352),
    // each cell = 2x2 conv outputs; mem1 lives in registers across 16 steps.
    float cur[6][4];
    float mem[6][4];
    for (int i = 0; i < 6; ++i) {
        int idx = tid + i * 256;
        if (idx >= KREAL) break;
        int c   = idx / 169;
        int rem = idx - c * 169;
        int py  = rem / 13;
        int px  = rem - py * 13;
        const float* wc = &ws[c * 9];
        #pragma unroll
        for (int dy = 0; dy < 2; ++dy)
        #pragma unroll
        for (int dx = 0; dx < 2; ++dx) {
            int r = 2 * py + dy, q = 2 * px + dx;
            float acc = 0.0f;
            #pragma unroll
            for (int ii = 0; ii < 3; ++ii)
            #pragma unroll
            for (int jj = 0; jj < 3; ++jj)
                acc = fmaf(xs[(r + ii) * 28 + (q + jj)], wc[ii * 3 + jj], acc);
            cur[i][dy * 2 + dx] = acc;
            mem[i][dy * 2 + dx] = 0.0f;
        }
    }

    for (int t = 0; t < NSTEP; ++t) {
        for (int i = 0; i < 6; ++i) {
            int idx = tid + i * 256;
            if (idx >= KREAL) break;
            float s = 0.0f;
            #pragma unroll
            for (int j = 0; j < 4; ++j) {
                float m  = BETA * mem[i][j] + cur[i][j];
                float sp = (m > THR) ? 1.0f : 0.0f;   // spike(mem - THR)
                mem[i][j] = m - sp * THR;             // subtract reset
                s += sp;
            }
            pool[t * KPAD + idx] = (_Float16)(0.25f * s);  // exact in f16
        }
    }
    __syncthreads();

    // ---- Phase 2: WMMA GEMM  cur2[m][n] = sum_k pool[m][k] * fc_w[n][k] ----
    const int lane = tid & 31;
    // Scalarize the wave index: loop trip count becomes SGPR-known, so the
    // kb loop compiles to scalar branches and EXEC stays all-1s for WMMA.
    const int wave = __builtin_amdgcn_readfirstlane(tid >> 5);
    const int mrow = lane & 15;             // A row (timestep) / B col (neuron)
    const int ksub = (lane >> 4) * 8;       // per-lane K sub-offset: 0 or 8

    v8f acc8;
    #pragma unroll
    for (int r = 0; r < 8; ++r) acc8[r] = 0.0f;

    const _Float16* pA = &pool[mrow * KPAD + ksub];
    const _Float16* pH = &fhi[mrow * KPAD + ksub];
    const _Float16* pL = &flo[mrow * KPAD + ksub];

    for (int kb = wave; kb < NKB; kb += 8) {
        const int k0 = kb * 32;
        // A fragment from LDS: halves {k0..k0+7, k0+16..k0+23} of row mrow
        v8h a0 = *(const v8h*)(pA + k0);
        v8h a1 = *(const v8h*)(pA + k0 + 16);
        v16h av = __builtin_shufflevector(a0, a1, 0,1,2,3,4,5,6,7,8,9,10,11,12,13,14,15);
        // B fragments (hi/lo weight planes) from global (L2-resident, 88 KB)
        v8h h0 = *(const v8h*)(pH + k0);
        v8h h1 = *(const v8h*)(pH + k0 + 16);
        v16h bh = __builtin_shufflevector(h0, h1, 0,1,2,3,4,5,6,7,8,9,10,11,12,13,14,15);
        v8h l0 = *(const v8h*)(pL + k0);
        v8h l1 = *(const v8h*)(pL + k0 + 16);
        v16h bl = __builtin_shufflevector(l0, l1, 0,1,2,3,4,5,6,7,8,9,10,11,12,13,14,15);

        acc8 = __builtin_amdgcn_wmma_f32_16x16x32_f16(false, av, false, bh,
                                                      (short)0, acc8, false, false);
        acc8 = __builtin_amdgcn_wmma_f32_16x16x32_f16(false, av, false, bl,
                                                      (short)0, acc8, false, false);
    }
    #pragma unroll
    for (int r = 0; r < 8; ++r) partial[wave * 256 + r * 32 + lane] = acc8[r];
    __syncthreads();

    // ---- Phase 3a: deterministic reduction of 8 wave partials ----
    {
        float s = 0.0f;
        #pragma unroll
        for (int w = 0; w < 8; ++w) s += partial[w * 256 + tid];
        int r = tid >> 5, L = tid & 31;
        int M = r + ((L >> 4) << 3);   // C/D layout: VGPR r, lanes 16-31 -> M+8
        int N = L & 15;
        cur2[M * 16 + N] = s;
    }
    __syncthreads();

    // ---- Phase 3b: LIF layer 2 recurrence + spike count ----
    if (tid < 10) {
        float m2 = 0.0f, cnt = 0.0f;
        #pragma unroll
        for (int t = 0; t < NSTEP; ++t) {
            m2 = BETA * m2 + cur2[t * 16 + tid];
            if (m2 > THR) { cnt += 1.0f; m2 -= THR; }
        }
        out[b * 10 + tid] = cnt;
    }
}

// ---------------------------------------------------------------------------
extern "C" void kernel_launch(void* const* d_in, const int* in_sizes, int n_in,
                              void* d_out, int out_size, void* d_ws, size_t ws_size,
                              hipStream_t stream) {
    (void)n_in; (void)out_size; (void)ws_size;
    const float* x      = (const float*)d_in[0];   // [B,1,28,28]
    const float* conv_w = (const float*)d_in[1];   // [8,1,3,3]
    const float* fc_w   = (const float*)d_in[2];   // [10,1352]
    float* out = (float*)d_out;                    // [B,10]

    const int B = in_sizes[0] / 784;

    _Float16* fhi = (_Float16*)d_ws;               // [16,1376] hi plane
    _Float16* flo = fhi + 16 * KPAD;               // [16,1376] lo plane

    snn_prep_fcw<<<(16 * KPAD + 255) / 256, 256, 0, stream>>>(fc_w, fhi, flo);
    snn_fused<<<B, 256, 0, stream>>>(x, conv_w, fhi, flo, out);
}
